// MultiDoubleStreamBlockLoraProcessor_38955353375467
// MI455X (gfx1250) — compile-verified, matrix operand order checked by reading, not run on hardware
//
#include <hip/hip_runtime.h>
#include <hip/hip_bf16.h>

// ---------------------------------------------------------------------------
// MI455X / gfx1250 implementation of MultiDoubleStreamBlockLoraProcessor.
// All matmuls use v_wmma_f32_16x16x32_bf16 (wave32 WMMA). fp32 data is
// converted to bf16 on the fly; accumulation is fp32.
// GEMM is software-pipelined; all hot global accesses use addrspace(1)
// (global_load/global_store, not flat) so LOADcnt/STOREcnt stay decoupled
// from DScnt and the LDS pipe.
// Workspace requirement: ~81 MB (see offsets in kernel_launch).
// ---------------------------------------------------------------------------

typedef __attribute__((ext_vector_type(16))) __bf16 v16bf;
typedef __attribute__((ext_vector_type(8)))  __bf16 v8bf;
typedef __attribute__((ext_vector_type(4)))  __bf16 v4bf;
typedef __attribute__((ext_vector_type(8)))  float  v8f;
typedef __attribute__((ext_vector_type(4)))  float  v4f;

#define DIMC   2048
#define SEQC   2176
#define IMGC   2048
#define TXTC   128
#define HEADSC 16
#define HDC    128
#define SBSC   1152          // mask row/col threshold (multiple of 32)
#define QTILES (SEQC / 16)   // 136

// --- forced-global (addrspace 1) access helpers: emit global_*, never flat_*
__device__ __forceinline__ v4f gload4(const float* p) {
  return *(const __attribute__((address_space(1))) v4f*)(unsigned long long)p;
}
__device__ __forceinline__ v8bf gload8bf(const __bf16* p) {
  return *(const __attribute__((address_space(1))) v8bf*)(unsigned long long)p;
}
__device__ __forceinline__ float gloadf(const float* p) {
  return *(const __attribute__((address_space(1))) float*)(unsigned long long)p;
}
__device__ __forceinline__ void gstoref(float* p, float v) {
  *(__attribute__((address_space(1))) float*)(unsigned long long)p = v;
}
__device__ __forceinline__ void gstorebf(__bf16* p, __bf16 v) {
  *(__attribute__((address_space(1))) __bf16*)(unsigned long long)p = v;
}

__device__ __forceinline__ v16bf cat16(v8bf lo, v8bf hi) {
  return __builtin_shufflevector(lo, hi, 0,1,2,3,4,5,6,7,8,9,10,11,12,13,14,15);
}

__device__ __forceinline__ v4bf cvt4(v4f v) {
  v4bf b;
  b[0] = (__bf16)v[0]; b[1] = (__bf16)v[1];
  b[2] = (__bf16)v[2]; b[3] = (__bf16)v[3];
  return b;
}

__device__ __forceinline__ v8f wmma_bf16(v16bf a, v16bf b, v8f c) {
  // D = A(16x32 bf16) * B(32x16 bf16) + C(16x16 f32)
  return __builtin_amdgcn_wmma_f32_16x16x32_bf16(
      /*neg_a=*/false, a, /*neg_b=*/false, b,
      /*c_mod=*/(short)0, c, /*reuse_a=*/false, /*reuse_b=*/false);
}

// ---------------------------------------------------------------------------
// Generic NT GEMM: C(MxN) = [C +] alpha * A(MxK) * B(NxK)^T, A/B fp32 in
// memory (converted to bf16 while staging to LDS), C fp32.
// Block tile 128x64, 8 waves of 16x64, K step 32. M%128==0, N%64==0, K%32==0.
// Software pipelined: global loads for chunk k+1 issued before the WMMA work
// of chunk k; fixed trip counts and hoisted pointers.
// ---------------------------------------------------------------------------
__global__ __launch_bounds__(256) void gemm_nt_bf16(
    const float* __restrict__ A, const float* __restrict__ B,
    float* __restrict__ C, int M, int N, int K,
    int lda, int ldb, int ldc, float alpha, int accum)
{
  __shared__ __align__(16) __bf16 sA[128 * 40];   // stride 40 (=80B) kills bank conflicts
  __shared__ __align__(16) __bf16 sB[64 * 40];

  const int tid  = threadIdx.x;
  const int wave = tid >> 5;
  const int lane = tid & 31;
  const int r    = lane & 15;      // row-in-tile (A) / col-in-tile (B)
  const int hf   = lane >> 4;      // 16-lane half selects K sub-chunk
  const int mbase = blockIdx.y * 128;
  const int nbase = blockIdx.x * 64;

  // Per-thread staging slots: 4 for A (128x32/256 threads, 4 floats each),
  // 2 for B (64x32/256 threads). Pointers bumped by +32 floats per K chunk.
  const float* aPtr[4];
  const float* bPtr[2];
  int aLds[4], bLds[2];
#pragma unroll
  for (int i = 0; i < 4; ++i) {
    int idx = tid + i * 256;              // 0..1023
    int row = idx >> 3;                   // 0..127
    int c4  = (idx & 7) << 2;             // 0,4,..,28
    aPtr[i] = A + (size_t)(mbase + row) * lda + c4;
    aLds[i] = row * 40 + c4;
  }
#pragma unroll
  for (int i = 0; i < 2; ++i) {
    int idx = tid + i * 256;              // 0..511
    int row = idx >> 3;                   // 0..63
    int c4  = (idx & 7) << 2;
    bPtr[i] = B + (size_t)(nbase + row) * ldb + c4;
    bLds[i] = row * 40 + c4;
  }

  v4f aReg[4], bReg[2];
#pragma unroll
  for (int i = 0; i < 4; ++i) { aReg[i] = gload4(aPtr[i]); aPtr[i] += 32; }
#pragma unroll
  for (int i = 0; i < 2; ++i) { bReg[i] = gload4(bPtr[i]); bPtr[i] += 32; }

  v8f acc[4];
#pragma unroll
  for (int ns = 0; ns < 4; ++ns)
#pragma unroll
    for (int j = 0; j < 8; ++j) acc[ns][j] = 0.0f;

  const int aFragOff = (wave * 16 + r) * 40 + hf * 8;

  for (int kb = 0; kb < K; kb += 32) {
    // commit in-register tile to LDS (fp32 -> bf16)
#pragma unroll
    for (int i = 0; i < 4; ++i) *(v4bf*)(sA + aLds[i]) = cvt4(aReg[i]);
#pragma unroll
    for (int i = 0; i < 2; ++i) *(v4bf*)(sB + bLds[i]) = cvt4(bReg[i]);
    __syncthreads();

    // prefetch next K chunk while this one is consumed by WMMAs
    if (kb + 32 < K) {
#pragma unroll
      for (int i = 0; i < 4; ++i) { aReg[i] = gload4(aPtr[i]); aPtr[i] += 32; }
#pragma unroll
      for (int i = 0; i < 2; ++i) { bReg[i] = gload4(bPtr[i]); bPtr[i] += 32; }
    }

    // A fragment: lane r holds row (wave*16+r); K halves hf*8 and 16+hf*8
    v8bf alo = *(const v8bf*)(sA + aFragOff);
    v8bf ahi = *(const v8bf*)(sA + aFragOff + 16);
    v16bf af = cat16(alo, ahi);

    // load all four B fragments, then issue WMMAs back-to-back
    v16bf bf4[4];
#pragma unroll
    for (int ns = 0; ns < 4; ++ns) {
      v8bf blo = *(const v8bf*)(sB + (ns * 16 + r) * 40 + hf * 8);
      v8bf bhi = *(const v8bf*)(sB + (ns * 16 + r) * 40 + 16 + hf * 8);
      bf4[ns] = cat16(blo, bhi);
    }
#pragma unroll
    for (int ns = 0; ns < 4; ++ns)
      acc[ns] = wmma_bf16(af, bf4[ns], acc[ns]);
    __syncthreads();
  }

  // Epilogue: VGPR j, lane -> row = j + hf*8, col = ns*16 + r
#pragma unroll
  for (int ns = 0; ns < 4; ++ns)
#pragma unroll
    for (int j = 0; j < 8; ++j) {
      int row = mbase + wave * 16 + j + hf * 8;
      int col = nbase + ns * 16 + r;
      float v = alpha * acc[ns][j];
      float* cp = C + (size_t)row * ldc + col;
      if (accum) gstoref(cp, gloadf(cp) + v);
      else       gstoref(cp, v);
    }
}

// ---------------------------------------------------------------------------
// Prep: RMS-norm(q,k)+gain, RoPE, fold 1/sqrt(HD) into q, emit bf16 Q/K and
// transposed bf16 V^T[h*HD+d][s]. One block per sequence row.
// ---------------------------------------------------------------------------
__global__ __launch_bounds__(256) void prep_kernel(
    const float* __restrict__ Qf, const float* __restrict__ Kf,
    const float* __restrict__ Vf,
    const float* __restrict__ g_q,  const float* __restrict__ g_k,
    const float* __restrict__ g_qt, const float* __restrict__ g_kt,
    const float* __restrict__ rope,
    __bf16* __restrict__ Qb, __bf16* __restrict__ Kb, __bf16* __restrict__ Vt)
{
  const int s   = blockIdx.x;
  const int tid = threadIdx.x;
  __shared__ float redq[256];
  __shared__ float redk[256];

  const float* q = Qf + (size_t)s * DIMC;
  const float* k = Kf + (size_t)s * DIMC;

  float ssq = 0.f, ssk = 0.f;
  for (int i = tid; i < DIMC; i += 256) {
    float a = gloadf(q + i); ssq += a * a;
    float b = gloadf(k + i); ssk += b * b;
  }
  redq[tid] = ssq; redk[tid] = ssk;
  __syncthreads();
  for (int off = 128; off > 0; off >>= 1) {
    if (tid < off) { redq[tid] += redq[tid + off]; redk[tid] += redk[tid + off]; }
    __syncthreads();
  }
  const float rq = rsqrtf(redq[0] * (1.0f / DIMC) + 1e-5f);
  const float rk = rsqrtf(redk[0] * (1.0f / DIMC) + 1e-5f);

  const float* gq = (s < IMGC) ? g_q : g_qt;
  const float* gk = (s < IMGC) ? g_k : g_kt;
  const float qscale = 0.08838834764831845f;   // 1/sqrt(HD)

  // RoPE over pairs: rope[(s*64 + i)*4 + {0..3}] = {cos,-sin,sin,cos}
  for (int p = tid; p < DIMC / 2; p += 256) {
    int i = p & 63;        // pair index within head (HD/2 = 64)
    int e = 2 * p;
    v4f rp = gload4(rope + ((size_t)s * 64 + i) * 4);
    float r0 = rp[0], r1 = rp[1], r2 = rp[2], r3 = rp[3];

    float qe = gloadf(q + e) * rq * gloadf(gq + e);
    float qo = gloadf(q + e + 1) * rq * gloadf(gq + e + 1);
    float ke = gloadf(k + e) * rk * gloadf(gk + e);
    float ko = gloadf(k + e + 1) * rk * gloadf(gk + e + 1);

    float qe2 = r0 * qe + r1 * qo, qo2 = r2 * qe + r3 * qo;
    float ke2 = r0 * ke + r1 * ko, ko2 = r2 * ke + r3 * ko;

    gstorebf(Qb + (size_t)s * DIMC + e,     (__bf16)(qe2 * qscale));
    gstorebf(Qb + (size_t)s * DIMC + e + 1, (__bf16)(qo2 * qscale));
    gstorebf(Kb + (size_t)s * DIMC + e,     (__bf16)ke2);
    gstorebf(Kb + (size_t)s * DIMC + e + 1, (__bf16)ko2);
  }
  for (int i = tid; i < DIMC; i += 256)
    gstorebf(Vt + (size_t)i * SEQC + s, (__bf16)gloadf(Vf + (size_t)s * DIMC + i));
}

// ---------------------------------------------------------------------------
// Flash attention. One wave per (head, 16-query tile); 4 waves per block.
// Mask = column range: rows >= 1152 start at key-chunk 36 (=1152/32).
// ---------------------------------------------------------------------------
__device__ __forceinline__ float rowmax16(float v) {
  v = fmaxf(v, __shfl_xor(v, 1));
  v = fmaxf(v, __shfl_xor(v, 2));
  v = fmaxf(v, __shfl_xor(v, 4));
  v = fmaxf(v, __shfl_xor(v, 8));
  return v;
}
__device__ __forceinline__ float rowsum16(float v) {
  v += __shfl_xor(v, 1);
  v += __shfl_xor(v, 2);
  v += __shfl_xor(v, 4);
  v += __shfl_xor(v, 8);
  return v;
}

__global__ __launch_bounds__(128) void attn_kernel(
    const __bf16* __restrict__ Qb, const __bf16* __restrict__ Kb,
    const __bf16* __restrict__ Vt, float* __restrict__ O)
{
  __shared__ __align__(16) __bf16 pb[4][16 * 40];   // per-wave P staging

  const int tid  = threadIdx.x;
  const int warp = tid >> 5;
  const int lane = tid & 31;
  const int r    = lane & 15;
  const int hf   = lane >> 4;

  const int job  = blockIdx.x * 4 + warp;           // 16 heads * 136 tiles
  const int head = job / QTILES;
  const int q0   = (job % QTILES) * 16;

  // Q fragments for all four 32-wide K chunks of HD=128 (kept resident)
  v16bf qf[4];
  const __bf16* qrow = Qb + (size_t)(q0 + r) * DIMC + head * HDC;
#pragma unroll
  for (int d = 0; d < 4; ++d) {
    v8bf lo = gload8bf(qrow + d * 32 + hf * 8);
    v8bf hi = gload8bf(qrow + d * 32 + 16 + hf * 8);
    qf[d] = cat16(lo, hi);
  }

  v8f o[8];
  float mrow[8], lrow[8];
#pragma unroll
  for (int ns = 0; ns < 8; ++ns)
#pragma unroll
    for (int j = 0; j < 8; ++j) o[ns][j] = 0.0f;
#pragma unroll
  for (int j = 0; j < 8; ++j) { mrow[j] = -1.0e30f; lrow[j] = 0.0f; }

  const int kc0 = (q0 >= SBSC) ? (SBSC / 32) : 0;

  for (int kc = kc0; kc < SEQC / 32; ++kc) {
    const int k0 = kc * 32;

    // ---- scores: two 16x16 tiles over this 32-key chunk.
    // Load the 4 K fragments of a tile first, then 4 back-to-back WMMAs.
    v8f s0, s1;
#pragma unroll
    for (int j = 0; j < 8; ++j) { s0[j] = 0.0f; s1[j] = 0.0f; }
    {
      v16bf kf[4];
      const __bf16* krow = Kb + (size_t)(k0 + r) * DIMC + head * HDC;
#pragma unroll
      for (int d = 0; d < 4; ++d) {
        v8bf lo = gload8bf(krow + d * 32 + hf * 8);
        v8bf hi = gload8bf(krow + d * 32 + 16 + hf * 8);
        kf[d] = cat16(lo, hi);
      }
#pragma unroll
      for (int d = 0; d < 4; ++d) s0 = wmma_bf16(qf[d], kf[d], s0);

      const __bf16* krow1 = Kb + (size_t)(k0 + 16 + r) * DIMC + head * HDC;
#pragma unroll
      for (int d = 0; d < 4; ++d) {
        v8bf lo = gload8bf(krow1 + d * 32 + hf * 8);
        v8bf hi = gload8bf(krow1 + d * 32 + 16 + hf * 8);
        kf[d] = cat16(lo, hi);
      }
#pragma unroll
      for (int d = 0; d < 4; ++d) s1 = wmma_bf16(qf[d], kf[d], s1);
    }

    // ---- online softmax (row = VGPR slot j across one 16-lane half) ----
#pragma unroll
    for (int j = 0; j < 8; ++j) {
      float smax = rowmax16(fmaxf(s0[j], s1[j]));
      float mnew = fmaxf(mrow[j], smax);
      float corr = __expf(mrow[j] - mnew);
      float p0 = __expf(s0[j] - mnew);
      float p1 = __expf(s1[j] - mnew);
      float rs = rowsum16(p0 + p1);
      lrow[j] = lrow[j] * corr + rs;
      mrow[j] = mnew;
#pragma unroll
      for (int ns = 0; ns < 8; ++ns) o[ns][j] *= corr;
      int m = j + hf * 8;
      pb[warp][m * 40 + r]      = (__bf16)p0;     // key t=0
      pb[warp][m * 40 + 16 + r] = (__bf16)p1;     // key t=1
    }

    // ---- P*V : re-read P as A-fragment, V^T gives contiguous B-fragments.
    // Load 4 V fragments, fire 4 WMMAs, twice.
    v8bf plo = *(const v8bf*)(&pb[warp][r * 40 + hf * 8]);
    v8bf phi = *(const v8bf*)(&pb[warp][r * 40 + 16 + hf * 8]);
    v16bf pf = cat16(plo, phi);
#pragma unroll
    for (int g = 0; g < 2; ++g) {
      v16bf vf4[4];
#pragma unroll
      for (int i = 0; i < 4; ++i) {
        int ns = g * 4 + i;
        const __bf16* vr = Vt + (size_t)(head * HDC + ns * 16 + r) * SEQC + k0;
        v8bf vlo = gload8bf(vr + hf * 8);
        v8bf vhi = gload8bf(vr + 16 + hf * 8);
        vf4[i] = cat16(vlo, vhi);
      }
#pragma unroll
      for (int i = 0; i < 4; ++i)
        o[g * 4 + i] = wmma_bf16(pf, vf4[i], o[g * 4 + i]);
    }
  }

  // ---- finalize: divide by row sum, store fp32 ----
#pragma unroll
  for (int j = 0; j < 8; ++j) {
    float inv = 1.0f / lrow[j];
    int m = q0 + j + hf * 8;
#pragma unroll
    for (int ns = 0; ns < 8; ++ns)
      gstoref(O + (size_t)m * DIMC + head * HDC + ns * 16 + r, o[ns][j] * inv);
  }
}

// ---------------------------------------------------------------------------
// Split H (2048x2048) into d_out img rows [0,1024) and cond rows [1024,2048)
// (cond region starts at element 1152*2048; text occupies [1024*2048,1152*2048)).
// ---------------------------------------------------------------------------
__global__ __launch_bounds__(256) void split_kernel(
    const float* __restrict__ H, float* __restrict__ out)
{
  size_t idx = (size_t)blockIdx.x * 256 + threadIdx.x;   // < 2048*2048
  int row = (int)(idx >> 11);
  float v = gloadf(H + idx);
  if (row < 1024) gstoref(out + idx, v);
  else            gstoref(out + (size_t)SBSC * DIMC + (idx - (size_t)1024 * DIMC), v);
}

// ---------------------------------------------------------------------------
extern "C" void kernel_launch(void* const* d_in, const int* in_sizes, int n_in,
                              void* d_out, int out_size, void* d_ws, size_t ws_size,
                              hipStream_t stream)
{
  const float* X    = (const float*)d_in[0];   // image tokens 2048x2048
  const float* T    = (const float*)d_in[1];   // text tokens   128x2048
  const float* rope = (const float*)d_in[2];
  const float* wq   = (const float*)d_in[3];
  const float* wk   = (const float*)d_in[4];
  const float* wv   = (const float*)d_in[5];
  const float* wo   = (const float*)d_in[6];
  const float* wqt  = (const float*)d_in[7];
  const float* wkt  = (const float*)d_in[8];
  const float* wvt  = (const float*)d_in[9];
  const float* wot  = (const float*)d_in[10];
  const float* g_q  = (const float*)d_in[11];
  const float* g_k  = (const float*)d_in[12];
  const float* g_qt = (const float*)d_in[13];
  const float* g_kt = (const float*)d_in[14];
  const float* qd   = (const float*)d_in[15];
  const float* qu   = (const float*)d_in[16];
  const float* kd   = (const float*)d_in[17];
  const float* ku   = (const float*)d_in[18];
  const float* vd   = (const float*)d_in[19];
  const float* vu   = (const float*)d_in[20];
  const float* pd   = (const float*)d_in[21];
  const float* pu   = (const float*)d_in[22];
  float* out = (float*)d_out;

  // Workspace carve-up (needs ~81 MB)
  char* w = (char*)d_ws;
  const size_t SZF  = (size_t)SEQC * DIMC * 4;   // 17,825,792
  const size_t SZB  = (size_t)SEQC * DIMC * 2;   //  8,912,896
  float*  Qf   = (float*)(w);
  float*  Kf   = (float*)(w + SZF);
  float*  Vf   = (float*)(w + 2 * SZF);
  __bf16* Qb   = (__bf16*)(w + 3 * SZF);
  __bf16* Kb   = (__bf16*)(w + 3 * SZF + SZB);
  __bf16* Vt   = (__bf16*)(w + 3 * SZF + 2 * SZB);
  float*  tbuf = (float*)(w + 3 * SZF + 3 * SZB); // 1024x64 fp32, reused
  float*  Attn = Qf;                               // alias: Qf dead after prep
  float*  H    = Kf;                               // alias: Kf dead after prep

  const dim3 blk(256);
  #define GGRID(M, N) dim3((N) / 64, (M) / 128)

  // ---- image QKV projections + LoRA (rank 64, rows [1024,2048)) ----
  gemm_nt_bf16<<<GGRID(IMGC, DIMC), blk, 0, stream>>>(X, wq, Qf, IMGC, DIMC, DIMC, DIMC, DIMC, DIMC, 1.f, 0);
  gemm_nt_bf16<<<GGRID(1024, 64),   blk, 0, stream>>>(X + (size_t)1024 * DIMC, qd, tbuf, 1024, 64, DIMC, DIMC, DIMC, 64, 1.f, 0);
  gemm_nt_bf16<<<GGRID(1024, DIMC), blk, 0, stream>>>(tbuf, qu, Qf + (size_t)1024 * DIMC, 1024, DIMC, 64, 64, 64, DIMC, 1.f, 1);

  gemm_nt_bf16<<<GGRID(IMGC, DIMC), blk, 0, stream>>>(X, wk, Kf, IMGC, DIMC, DIMC, DIMC, DIMC, DIMC, 1.f, 0);
  gemm_nt_bf16<<<GGRID(1024, 64),   blk, 0, stream>>>(X + (size_t)1024 * DIMC, kd, tbuf, 1024, 64, DIMC, DIMC, DIMC, 64, 1.f, 0);
  gemm_nt_bf16<<<GGRID(1024, DIMC), blk, 0, stream>>>(tbuf, ku, Kf + (size_t)1024 * DIMC, 1024, DIMC, 64, 64, 64, DIMC, 1.f, 1);

  gemm_nt_bf16<<<GGRID(IMGC, DIMC), blk, 0, stream>>>(X, wv, Vf, IMGC, DIMC, DIMC, DIMC, DIMC, DIMC, 1.f, 0);
  gemm_nt_bf16<<<GGRID(1024, 64),   blk, 0, stream>>>(X + (size_t)1024 * DIMC, vd, tbuf, 1024, 64, DIMC, DIMC, DIMC, 64, 1.f, 0);
  gemm_nt_bf16<<<GGRID(1024, DIMC), blk, 0, stream>>>(tbuf, vu, Vf + (size_t)1024 * DIMC, 1024, DIMC, 64, 64, 64, DIMC, 1.f, 1);

  // ---- text QKV projections into rows [2048, 2176) ----
  gemm_nt_bf16<<<GGRID(TXTC, DIMC), blk, 0, stream>>>(T, wqt, Qf + (size_t)IMGC * DIMC, TXTC, DIMC, DIMC, DIMC, DIMC, DIMC, 1.f, 0);
  gemm_nt_bf16<<<GGRID(TXTC, DIMC), blk, 0, stream>>>(T, wkt, Kf + (size_t)IMGC * DIMC, TXTC, DIMC, DIMC, DIMC, DIMC, DIMC, 1.f, 0);
  gemm_nt_bf16<<<GGRID(TXTC, DIMC), blk, 0, stream>>>(T, wvt, Vf + (size_t)IMGC * DIMC, TXTC, DIMC, DIMC, DIMC, DIMC, DIMC, 1.f, 0);

  // ---- RMS + gain + RoPE + bf16 + V^T ----
  prep_kernel<<<SEQC, blk, 0, stream>>>(Qf, Kf, Vf, g_q, g_k, g_qt, g_kt, rope, Qb, Kb, Vt);

  // ---- flash attention (writes Attn, aliases Qf) ----
  attn_kernel<<<(HEADSC * QTILES) / 4, dim3(128), 0, stream>>>(Qb, Kb, Vt, Attn);

  // ---- output projections ----
  gemm_nt_bf16<<<GGRID(IMGC, DIMC), blk, 0, stream>>>(Attn, wo, H, IMGC, DIMC, DIMC, DIMC, DIMC, DIMC, 1.f, 0);
  gemm_nt_bf16<<<GGRID(TXTC, DIMC), blk, 0, stream>>>(Attn + (size_t)IMGC * DIMC, wot, out + (size_t)1024 * DIMC, TXTC, DIMC, DIMC, DIMC, DIMC, DIMC, 1.f, 0);

  // ---- p-LoRA on post-projection h_i (masked rows [1024,2048)) ----
  gemm_nt_bf16<<<GGRID(1024, 64),   blk, 0, stream>>>(H + (size_t)1024 * DIMC, pd, tbuf, 1024, 64, DIMC, DIMC, DIMC, 64, 1.f, 0);
  gemm_nt_bf16<<<GGRID(1024, DIMC), blk, 0, stream>>>(tbuf, pu, H + (size_t)1024 * DIMC, 1024, DIMC, 64, 64, 64, DIMC, 1.f, 1);

  // ---- split h_i into (h_img, cond_hidden) regions of d_out ----
  split_kernel<<<(IMGC * DIMC) / 256, blk, 0, stream>>>(H, out);
}
